// TSNStacked3_86225763435192
// MI455X (gfx1250) — matrix operand order repaired and checked by reading, not compiled, not run on hardware
//
#include <hip/hip_runtime.h>

// ---------------------------------------------------------------------------
// Problem constants (from the reference)
// ---------------------------------------------------------------------------
constexpr int Bc   = 4;
constexpr int Tc   = 12;
constexpr int NN   = 2048;
constexpr int Fc   = 11;
constexpr int Hc   = 256;
constexpr int LFc  = 64;
constexpr int Ec   = 16384;
constexpr int HORc = 12;
constexpr int GRc  = 16;
constexpr int BT   = Bc * Tc;               // 48
constexpr int MFULL = BT * NN;              // 98304 rows
constexpr int CMAX = Hc + LFc + 4 * GRc;    // 384
constexpr int BNH  = Bc * NN * Hc;          // 2097152 (one LIF state tensor)

// Vector types
typedef __attribute__((ext_vector_type(16))) _Float16 v16h;
typedef __attribute__((ext_vector_type(8)))  _Float16 h8;   // 16B, natural align 16
typedef __attribute__((ext_vector_type(8)))  float    v8f;
typedef __attribute__((ext_vector_type(4)))  float    f4;

// Epilogue flags
constexpr int EPI_RELU = 1;
constexpr int EPI_RES  = 2;
constexpr int EPI_BIAS = 4;
constexpr int EPI_NODE = 8;

// ---------------------------------------------------------------------------
// WMMA GEMM: C[MxN] = epi(A[MxK] @ B[KxN]) ; fp32 storage, f16 tiles in LDS,
// v_wmma_f32_16x16x32_f16 accumulate. Block tile 128(M) x 32(N), 8 waves,
// each wave: one A fragment, two N-tiles (2 WMMA / K-chunk).
//
// PRECONDITIONS (guaranteed by the host driver below):
//   M % 128 == 0, K % 32 == 0,
//   A rows have >= K readable floats (zero beyond logical K),
//   B has >= ceil32 columns readable for every staged tile (padded weights).
// Only the epilogue guards gcol < N.
// ---------------------------------------------------------------------------
constexpr int BM = 128;
constexpr int BK = 32;
constexpr int BN = 32;
constexpr int LDA_S = 40;   // halves; 80B row stride (16B-aligned, bank-skewed)

template <int EPI>
__global__ __launch_bounds__(256) void wmma_gemm(
    const float* __restrict__ A, int lda,
    const float* __restrict__ Bw, int ldb,
    const float* __restrict__ bias,
    const float* __restrict__ res, int ldres,
    const float* __restrict__ nodeemb,
    float* __restrict__ C, int ldc,
    int M, int N, int K)
{
    __shared__ _Float16 As[BM * LDA_S];            // 10240 B
    __shared__ _Float16 Bs[2 * BN * 16];           // [kgroup][col][k] 2048 B

    const int tid  = threadIdx.x;
    const int wave = tid >> 5;
    const int lane = tid & 31;
    const int rl   = lane & 15;
    const int hi   = lane >> 4;

    const int rowBlock = blockIdx.y * BM;
    const int colBlock = blockIdx.x * BN;

    // A-stage: 256 threads x 16 floats (4 x b128) = 128x32 tile
    const int arow  = tid >> 1;
    const int acolb = (tid & 1) * 16;
    // B-stage: 256 threads x 4 floats (1 x b128) = 32x32 tile
    const int brow  = tid >> 3;          // k within chunk
    const int bcol4 = (tid & 7) * 4;
    const int bkg   = brow >> 4;
    const int bkr   = brow & 15;

    v8f acc0 = {};
    v8f acc1 = {};

    const float* aRowPtr = A + (size_t)(rowBlock + arow) * lda + acolb;

    for (int k0 = 0; k0 < K; k0 += BK) {
        // ---- stage A (branch-free, 4x global_load_b128) ----
        {
            const f4* ap = (const f4*)(aRowPtr + k0);
            _Float16* sp = &As[arow * LDA_S + acolb];
#pragma unroll
            for (int u = 0; u < 4; ++u) {
                f4 v = ap[u];
#pragma unroll
                for (int c2 = 0; c2 < 4; ++c2)
                    sp[u * 4 + c2] = (_Float16)v[c2];
            }
        }
        // ---- stage B transposed into fragment-ready layout ----
        {
            const f4* bp = (const f4*)(Bw + (size_t)(k0 + brow) * ldb + colBlock + bcol4);
            f4 v = bp[0];
#pragma unroll
            for (int c2 = 0; c2 < 4; ++c2)
                Bs[(bkg * BN + bcol4 + c2) * 16 + bkr] = (_Float16)v[c2];
        }
        __syncthreads();

        // speculative prefetch of next A chunk (global_prefetch_b8)
        if (k0 + BK < K) __builtin_prefetch(aRowPtr + k0 + BK, 0, 3);

        // ---- A fragment: 2x ds_load_b128 (contiguous k = 16*hi + s, matches B) ----
        v16h af;
        {
            const _Float16* arp = &As[(wave * 16 + rl) * LDA_S + hi * 16];
            h8 lo = *(const h8*)(arp);
            h8 hi8 = *(const h8*)(arp + 8);
#pragma unroll
            for (int j = 0; j < 8; ++j) { af[j] = lo[j]; af[8 + j] = hi8[j]; }
        }
        // ---- B fragments: contiguous 16 halves per lane per N-tile ----
        v16h bf0, bf1;
        {
            const h8* brp0 = (const h8*)&Bs[(hi * BN + rl) * 16];
            h8 b0 = brp0[0], b1 = brp0[1];
            const h8* brp1 = (const h8*)&Bs[(hi * BN + 16 + rl) * 16];
            h8 b2 = brp1[0], b3 = brp1[1];
#pragma unroll
            for (int j = 0; j < 8; ++j) {
                bf0[j] = b0[j]; bf0[8 + j] = b1[j];
                bf1[j] = b2[j]; bf1[8 + j] = b3[j];
            }
        }

        acc0 = __builtin_amdgcn_wmma_f32_16x16x32_f16(false, af, false, bf0,
                                                      (short)0, acc0, false, false);
        acc1 = __builtin_amdgcn_wmma_f32_16x16x32_f16(false, af, false, bf1,
                                                      (short)0, acc1, false, false);
        __syncthreads();
    }

    // ---- epilogue (C/D layout: VGPR r -> M=r+8*hi, N=lane&15) ----
#pragma unroll
    for (int ct = 0; ct < 2; ++ct) {
        const int gcol = colBlock + ct * 16 + rl;
        if (gcol < N) {
#pragma unroll
            for (int r = 0; r < 8; ++r) {
                const int grow = rowBlock + wave * 16 + hi * 8 + r;
                float v = ct ? acc1[r] : acc0[r];
                if constexpr (EPI & EPI_BIAS) v += bias[gcol];
                if constexpr (EPI & EPI_RES)  v += res[(size_t)grow * ldres + gcol];
                if constexpr (EPI & EPI_NODE) v += nodeemb[(size_t)(grow & (NN - 1)) * Hc + gcol];
                if constexpr (EPI & EPI_RELU) v = v > 0.0f ? v : 0.0f;
                C[(size_t)grow * ldc + gcol] = v;
            }
        }
    }
}

static void launch_gemm(hipStream_t s, int epi,
                        const float* A, int lda,
                        const float* Bw, int ldb,
                        const float* bias,
                        const float* res, int ldres,
                        const float* nodeemb,
                        float* C, int ldc,
                        int M, int N, int K)
{
    dim3 grid((N + BN - 1) / BN, M / BM);
    dim3 blk(256);
    switch (epi) {
    case EPI_BIAS:
        wmma_gemm<EPI_BIAS><<<grid, blk, 0, s>>>(A, lda, Bw, ldb, bias, res, ldres, nodeemb, C, ldc, M, N, K);
        break;
    case EPI_RELU:
        wmma_gemm<EPI_RELU><<<grid, blk, 0, s>>>(A, lda, Bw, ldb, bias, res, ldres, nodeemb, C, ldc, M, N, K);
        break;
    case EPI_BIAS | EPI_RES:
        wmma_gemm<EPI_BIAS | EPI_RES><<<grid, blk, 0, s>>>(A, lda, Bw, ldb, bias, res, ldres, nodeemb, C, ldc, M, N, K);
        break;
    case EPI_BIAS | EPI_RELU:
        wmma_gemm<EPI_BIAS | EPI_RELU><<<grid, blk, 0, s>>>(A, lda, Bw, ldb, bias, res, ldres, nodeemb, C, ldc, M, N, K);
        break;
    case EPI_BIAS | EPI_NODE:
        wmma_gemm<EPI_BIAS | EPI_NODE><<<grid, blk, 0, s>>>(A, lda, Bw, ldb, bias, res, ldres, nodeemb, C, ldc, M, N, K);
        break;
    default: break;
    }
}

// ---------------------------------------------------------------------------
// Packing / utility kernels
// ---------------------------------------------------------------------------
__global__ void zero_kernel(float* p, size_t n)
{
    size_t i = (size_t)blockIdx.x * blockDim.x + threadIdx.x;
    size_t stride = (size_t)gridDim.x * blockDim.x;
    for (; i < n; i += stride) p[i] = 0.0f;
}

__global__ void relu_inplace(float* p, size_t n)
{
    size_t i = (size_t)blockIdx.x * blockDim.x + threadIdx.x;
    if (i < n) { float v = p[i]; p[i] = v > 0.0f ? v : 0.0f; }
}

// zero-padded weight repack: (K x N) -> (Kp x Np)
__global__ void pack_w(const float* __restrict__ w, float* __restrict__ wp,
                       int K, int N, int Kp, int Np)
{
    int idx = blockIdx.x * blockDim.x + threadIdx.x;
    if (idx >= Kp * Np) return;
    int c = idx % Np;
    int r = idx / Np;
    wp[idx] = (r < K && c < N) ? w[(size_t)r * N + c] : 0.0f;
}

// x (MFULL x 11) -> xpad (MFULL x 32, zero-padded)
__global__ void pack_x(const float* __restrict__ x, float* __restrict__ xp)
{
    size_t idx = (size_t)blockIdx.x * blockDim.x + threadIdx.x;
    if (idx >= (size_t)MFULL * 32) return;
    int c = (int)(idx & 31);
    size_t r = idx >> 5;
    xp[idx] = (c < Fc) ? x[r * Fc + c] : 0.0f;
}

// edge-weight MLP: 1 -> 32 (relu) -> 1
__global__ void ew_mlp(const float* __restrict__ ew,
                       const float* __restrict__ w1, const float* __restrict__ b1,
                       const float* __restrict__ w2, const float* __restrict__ b2,
                       float* __restrict__ out, int E)
{
    int e = blockIdx.x * blockDim.x + threadIdx.x;
    if (e >= E) return;
    float v = ew[e];
    float acc = b2[0];
#pragma unroll
    for (int j = 0; j < 32; ++j) {
        float h = v * w1[j] + b1[j];
        h = h > 0.0f ? h : 0.0f;
        acc += h * w2[j];
    }
    out[e] = acc;
}

// hbuf[:, :, 0:256] = out_acc ; hbuf[:, :, 256:320] = lw[n]
__global__ void build_concat(const float* __restrict__ outa,
                             const float* __restrict__ lw,
                             float* __restrict__ hbuf)
{
    size_t total = (size_t)MFULL * (Hc + LFc);
    size_t idx = (size_t)blockIdx.x * blockDim.x + threadIdx.x;
    if (idx >= total) return;
    int ch = (int)(idx % (Hc + LFc));
    size_t rn = idx / (Hc + LFc);
    int n = (int)(rn % NN);
    float v = (ch < Hc) ? outa[rn * Hc + ch] : lw[(size_t)n * LFc + (ch - Hc)];
    hbuf[rn * CMAX + ch] = v;
}

// message passing: agg[bt,dst,0:C] += ew[e] * h[bt,src,0:C]
__global__ void graph_scatter(const float* __restrict__ hbuf,
                              const int* __restrict__ src,
                              const int* __restrict__ dst,
                              const float* __restrict__ ew,
                              float* __restrict__ agg, int C)
{
    const int e  = blockIdx.x;
    const int bt = blockIdx.y;
    const int s  = src[e];
    const int d  = dst[e];
    const float w = ew[e];
    const float* hp = hbuf + ((size_t)bt * NN + s) * CMAX;
    float*       ap = agg  + ((size_t)bt * NN + d) * CMAX;
    for (int ch = threadIdx.x; ch < C; ch += blockDim.x)
        atomicAdd(&ap[ch], w * hp[ch]);
}

// LayerNorm over last dim (H=256), one row per 256-thread block (wave32 shuffles)
__global__ void layernorm(float* __restrict__ h,
                          const float* __restrict__ g,
                          const float* __restrict__ b)
{
    __shared__ float red[8];
    const size_t row = blockIdx.x;
    const int tid  = threadIdx.x;
    const int lane = tid & 31;
    const int wave = tid >> 5;

    float v = h[row * Hc + tid];

    float s = v;
#pragma unroll
    for (int o = 16; o > 0; o >>= 1) s += __shfl_xor(s, o, 32);
    if (lane == 0) red[wave] = s;
    __syncthreads();
    float tot = 0.0f;
#pragma unroll
    for (int i = 0; i < 8; ++i) tot += red[i];
    const float mu = tot * (1.0f / Hc);
    const float dv = v - mu;

    float q = dv * dv;
#pragma unroll
    for (int o = 16; o > 0; o >>= 1) q += __shfl_xor(q, o, 32);
    __syncthreads();
    if (lane == 0) red[wave] = q;
    __syncthreads();
    float qt = 0.0f;
#pragma unroll
    for (int i = 0; i < 8; ++i) qt += red[i];
    const float var = qt * (1.0f / Hc);

    h[row * Hc + tid] = dv * rsqrtf(var + 1e-5f) * g[tid] + b[tid];
}

// gather timestep t: xt[(b*N+n)*H+c] = hH[((b*T+t)*N+n)*H+c]
__global__ void pack_time(const float* __restrict__ hH, float* __restrict__ xt, int t)
{
    int idx = blockIdx.x * blockDim.x + threadIdx.x;
    if (idx >= BNH) return;
    int c = idx & (Hc - 1);
    int n = (idx >> 8) & (NN - 1);
    int b = idx >> 19;
    xt[idx] = hH[(((size_t)(b * Tc + t) * NN + n) * Hc) + c];
}

// LIF neuron update
__global__ void lif_update(const float* __restrict__ I,
                           float* __restrict__ syn,
                           float* __restrict__ mem,
                           float* __restrict__ spk, int n)
{
    int i = blockIdx.x * blockDim.x + threadIdx.x;
    if (i >= n) return;
    float s = 0.9f * syn[i] + I[i];
    float m = 0.8f * mem[i] + s;
    float sp = (m - 1.0f) > 0.0f ? 1.0f : 0.0f;
    m -= sp;
    syn[i] = s; mem[i] = m; spk[i] = sp;
}

// (B,N,HOR,F) -> (B,HOR,N,F)
__global__ void write_out(const float* __restrict__ d2, float* __restrict__ out)
{
    int idx = blockIdx.x * blockDim.x + threadIdx.x;
    const int total = Bc * NN * HORc * Fc;
    if (idx >= total) return;
    int f = idx % Fc;   int r = idx / Fc;
    int hor = r % HORc; r /= HORc;
    int n = r % NN;     int b = r / NN;
    out[(((size_t)(b * HORc + hor) * NN + n) * Fc) + f] =
        d2[((size_t)(b * NN + n) * (HORc * Fc)) + hor * Fc + f];
}

// ---------------------------------------------------------------------------
// Host orchestration. Input leaf order (jax tree_leaves, sorted dict keys):
//  0:x 1:edge_index 2:edge_weight
//  blocks[0]: 3..6 conv_ws, 7 ln_b, 8 ln_g, 9 lw, 10 proj_b, 11 proj_w,
//             12 skip_b, 13 skip_w ; blocks[1]: 14..24 same layout
//  25 dec_b1, 26 dec_b2, 27 dec_w1, 28 dec_w2, 29 enc_b, 30 enc_w,
//  31 ew_b1, 32 ew_b2, 33 ew_w1, 34 ew_w2, 35 node_emb,
//  36..38 syn_bs, 39..41 syn_ws
// ---------------------------------------------------------------------------
extern "C" void kernel_launch(void* const* d_in, const int* in_sizes, int n_in,
                              void* d_out, int out_size, void* d_ws, size_t ws_size,
                              hipStream_t stream)
{
    const float* x   = (const float*)d_in[0];
    const int*   ei  = (const int*)d_in[1];
    const float* ewt = (const float*)d_in[2];
    const int* src = ei;
    const int* dst = ei + Ec;

    const int blkBase[2] = {3, 14};
    const float* conv_w[2][4];
    const float* ln_b[2];  const float* ln_g[2];  const float* lw[2];
    const float* proj_b[2]; const float* proj_w[2];
    const float* skip_b[2]; const float* skip_w[2];
    for (int k = 0; k < 2; ++k) {
        int bs = blkBase[k];
        for (int j = 0; j < 4; ++j) conv_w[k][j] = (const float*)d_in[bs + j];
        ln_b[k]   = (const float*)d_in[bs + 4];
        ln_g[k]   = (const float*)d_in[bs + 5];
        lw[k]     = (const float*)d_in[bs + 6];
        proj_b[k] = (const float*)d_in[bs + 7];
        proj_w[k] = (const float*)d_in[bs + 8];
        skip_b[k] = (const float*)d_in[bs + 9];
        skip_w[k] = (const float*)d_in[bs + 10];
    }
    const float* dec_b1 = (const float*)d_in[25];
    const float* dec_b2 = (const float*)d_in[26];
    const float* dec_w1 = (const float*)d_in[27];
    const float* dec_w2 = (const float*)d_in[28];
    const float* enc_b  = (const float*)d_in[29];
    const float* enc_w  = (const float*)d_in[30];
    const float* ew_b1  = (const float*)d_in[31];
    const float* ew_b2  = (const float*)d_in[32];
    const float* ew_w1  = (const float*)d_in[33];
    const float* ew_w2  = (const float*)d_in[34];
    const float* node_emb = (const float*)d_in[35];
    const float* syn_b[3] = {(const float*)d_in[36], (const float*)d_in[37], (const float*)d_in[38]};
    const float* syn_w[3] = {(const float*)d_in[39], (const float*)d_in[40], (const float*)d_in[41]};

    // ---- workspace carve (floats; every offset stays 16B aligned) ----
    float* W = (float*)d_ws;
    size_t o = 0;
    float* ew_out = W + o; o += Ec;
    float* hH     = W + o; o += (size_t)MFULL * Hc;
    float* outa   = W + o; o += (size_t)MFULL * Hc;
    float* hbuf   = W + o; o += (size_t)MFULL * CMAX;
    float* agg    = W + o; o += (size_t)MFULL * CMAX;
    float* xt     = W + o; o += BNH;
    float* Ibuf   = W + o; o += BNH;
    float* synb   = W + o; o += (size_t)3 * BNH;
    float* memb   = W + o; o += (size_t)3 * BNH;
    float* spk    = W + o; o += BNH;
    float* dbuf1  = W + o; o += (size_t)Bc * NN * 4 * Hc;
    float* dbuf2  = W + o; o += (size_t)Bc * NN * HORc * Fc;
    float* xpad   = W + o; o += (size_t)MFULL * 32;          // x zero-padded to K=32
    float* encwP  = W + o; o += (size_t)32 * Hc;             // enc_w (32x256)
    float* convwP = W + o; o += (size_t)2 * 4 * CMAX * 32;   // 8 x (384x32)
    float* decw2P = W + o; o += (size_t)4 * Hc * 160;        // dec_w2 (1024x160)
    (void)ws_size; (void)n_in; (void)in_sizes; (void)out_size;

    // 0) pad ragged operands once (removes all K/N guards from GEMM hot loop)
    pack_x<<<(unsigned)(((size_t)MFULL * 32 + 255) / 256), 256, 0, stream>>>(x, xpad);
    pack_w<<<(32 * Hc + 255) / 256, 256, 0, stream>>>(enc_w, encwP, Fc, Hc, 32, Hc);
    for (int k = 0; k < 2; ++k) {
        int c = Hc + LFc;
        for (int j = 0; j < 4; ++j) {
            float* dstw = convwP + (size_t)(k * 4 + j) * CMAX * 32;
            pack_w<<<(CMAX * 32 + 255) / 256, 256, 0, stream>>>(conv_w[k][j], dstw,
                                                                c, GRc, CMAX, 32);
            c += GRc;
        }
    }
    pack_w<<<(4 * Hc * 160 + 255) / 256, 256, 0, stream>>>(dec_w2, decw2P,
                                                           4 * Hc, HORc * Fc, 4 * Hc, 160);

    // 1) edge-weight MLP
    ew_mlp<<<(Ec + 255) / 256, 256, 0, stream>>>(ewt, ew_w1, ew_b1, ew_w2, ew_b2, ew_out, Ec);

    // 2) encoder: hH = x @ enc_w + enc_b + node_emb[n]   (K padded to 32)
    launch_gemm(stream, EPI_BIAS | EPI_NODE, xpad, 32, encwP, Hc, enc_b,
                nullptr, 0, node_emb, hH, Hc, MFULL, Hc, 32);

    // 3) skip accumulator = 0
    zero_kernel<<<4096, 256, 0, stream>>>(outa, (size_t)MFULL * Hc);

    // 4) graph blocks
    for (int k = 0; k < 2; ++k) {
        launch_gemm(stream, EPI_BIAS | EPI_RES, hH, Hc, skip_w[k], Hc, skip_b[k],
                    outa, Hc, nullptr, outa, Hc, MFULL, Hc, Hc);
        {
            size_t total = (size_t)MFULL * (Hc + LFc);
            build_concat<<<(unsigned)((total + 255) / 256), 256, 0, stream>>>(outa, lw[k], hbuf);
        }
        int c = Hc + LFc;
        for (int j = 0; j < 4; ++j) {
            zero_kernel<<<4096, 256, 0, stream>>>(agg, (size_t)MFULL * CMAX);
            graph_scatter<<<dim3(Ec, BT), 128, 0, stream>>>(hbuf, src, dst, ew_out, agg, c);
            const float* dstw = convwP + (size_t)(k * 4 + j) * CMAX * 32;
            int kpad = (c + 31) & ~31;       // agg is zero beyond c, weights zero-padded
            launch_gemm(stream, EPI_RELU, agg, CMAX, dstw, 32, nullptr,
                        nullptr, 0, nullptr, hbuf + c, CMAX, MFULL, GRc, kpad);
            c += GRc;
        }
        launch_gemm(stream, EPI_BIAS, hbuf, CMAX, proj_w[k], Hc, proj_b[k],
                    nullptr, 0, nullptr, hH, Hc, MFULL, Hc, CMAX);
        layernorm<<<MFULL, 256, 0, stream>>>(hH, ln_g[k], ln_b[k]);
    }

    // 5) LIF recurrence over T
    zero_kernel<<<4096, 256, 0, stream>>>(synb, (size_t)6 * BNH); // syn+mem contiguous
    for (int t = 0; t < Tc; ++t) {
        pack_time<<<(BNH + 255) / 256, 256, 0, stream>>>(hH, xt, t);
        const float* inp = xt;
        for (int l = 0; l < 3; ++l) {
            launch_gemm(stream, EPI_BIAS, inp, Hc, syn_w[l], Hc, syn_b[l],
                        nullptr, 0, nullptr, Ibuf, Hc, Bc * NN, Hc, Hc);
            lif_update<<<(BNH + 255) / 256, 256, 0, stream>>>(
                Ibuf, synb + (size_t)l * BNH, memb + (size_t)l * BNH, spk, BNH);
            inp = spk;
        }
    }

    // 6) decoder
    relu_inplace<<<(BNH + 255) / 256, 256, 0, stream>>>(spk, BNH);
    launch_gemm(stream, EPI_BIAS | EPI_RELU, spk, Hc, dec_w1, 4 * Hc, dec_b1,
                nullptr, 0, nullptr, dbuf1, 4 * Hc, Bc * NN, 4 * Hc, Hc);
    launch_gemm(stream, EPI_BIAS, dbuf1, 4 * Hc, decw2P, 160, dec_b2,
                nullptr, 0, nullptr, dbuf2, HORc * Fc, Bc * NN, HORc * Fc, 4 * Hc);

    // 7) (B,N,HOR,F) -> (B,HOR,N,F)
    {
        const int total = Bc * NN * HORc * Fc;
        write_out<<<(total + 255) / 256, 256, 0, stream>>>(dbuf2, (float*)d_out);
    }
}
